// GemmaAttention_24953759990319
// MI455X (gfx1250) — matrix operand order repaired and checked by reading, not compiled, not run on hardware
//
#include <hip/hip_runtime.h>

// ---------------------------------------------------------------------------
// Problem constants (match reference)
// ---------------------------------------------------------------------------
#define B_   2
#define S_   2048
#define D_   2048
#define H_   8
#define HKV_ 1
#define HD_  256
#define BSR  (B_ * S_)        // 4096 total rows

typedef __attribute__((ext_vector_type(16))) __bf16 v16bf;
typedef __attribute__((ext_vector_type(8)))  float  v8f;
typedef unsigned int       u32;
typedef unsigned short     u16;
typedef unsigned long long u64;

union FragBF { v16bf v; u32 u[8]; };

// f32 -> bf16 bits, round-to-nearest-even
__device__ __forceinline__ u16 f2bf(float f) {
    u32 u = __float_as_uint(f);
    u += 0x7FFFu + ((u >> 16) & 1u);
    return (u16)(u >> 16);
}
__device__ __forceinline__ float bf2f(u16 h) {
    return __uint_as_float(((u32)h) << 16);
}

// ---------------------------------------------------------------------------
// gfx1250 async global->LDS copy (ASYNCcnt-tracked) via inline asm.
// VDST operand = LDS byte offset (low 32 bits of generic LDS pointer per the
// aperture scheme), VADDR = 64-bit global address. Fallback: sync VGPR copy.
// ---------------------------------------------------------------------------
__device__ __forceinline__ void copy_b128_to_lds(const void* gsrc, void* ldst) {
#if defined(__gfx1250__)
    asm volatile("global_load_async_to_lds_b128 %0, %1, off"
                 :: "v"((u32)(u64)ldst), "v"(gsrc)
                 : "memory");
#else
    *(uint4*)ldst = *(const uint4*)gsrc;
#endif
}

template <int N> __device__ __forceinline__ void wait_async() {
#if defined(__gfx1250__)
    if constexpr (N == 0)      asm volatile("s_wait_asynccnt 0x0" ::: "memory");
    else if constexpr (N == 2) asm volatile("s_wait_asynccnt 0x2" ::: "memory");
    else                       asm volatile("s_wait_asynccnt 0x4" ::: "memory");
#endif
}

// ---------------------------------------------------------------------------
// NT GEMM: C[M,N] = scale * A[M,K] x B[N,K]^T
//  A: bf16 (async-staged) or f32 (float4-load + pack)    [A_BF16]
//  B: always bf16 (async-staged, rows are N, contiguous K)
//  C: f32, bf16, or bf16-transposed (C^T[N,M], b128 stores) [OUT_BF16/OUT_TRANS]
// Tile 128x128, K-step 32, double-buffered LDS. 8 wave32; each wave: 64x32
// region = 4x2 WMMA tiles -> 8 x v_wmma_f32_16x16x32_bf16 per K-step.
// All launch shapes are exact multiples of the tile sizes.
// ---------------------------------------------------------------------------
template <bool A_BF16, bool OUT_BF16, bool OUT_TRANS>
__global__ __launch_bounds__(256)
void gemm_nt(const void* __restrict__ Ap, const u16* __restrict__ Bp,
             void* __restrict__ Cp,
             int M, int N, int K, int lda, int ldb, int ldc, float scale)
{
    __shared__ alignas(16) u16 sA[2][128][40];   // [buf][m][k] (+8 pad; rows stay 16B-aligned)
    __shared__ alignas(16) u16 sB[2][128][40];   // [buf][n][k]

    const int m0   = blockIdx.y * 128;
    const int n0   = blockIdx.x * 128;
    const int tid  = threadIdx.x;
    const int wid  = tid >> 5;
    const int lane = tid & 31;
    const int half = lane >> 4;
    const int l16  = lane & 15;
    const int wm   = (wid >> 2) * 64;
    const int wn   = (wid & 3) * 32;

    const u16*   Abf = (const u16*)Ap;
    const float* Af  = (const float*)Ap;

    v8f acc[4][2];
    #pragma unroll
    for (int mi = 0; mi < 4; ++mi)
        #pragma unroll
        for (int ni = 0; ni < 2; ++ni) { v8f z = {}; acc[mi][ni] = z; }

    // ---- stage one 32-wide K-tile into LDS buffer `buf` ----
    auto stage = [&](int buf, int k0) {
        if (A_BF16) {
            #pragma unroll
            for (int i = tid; i < 512; i += 256) {           // 128 rows x 4 chunks(8 bf16)
                int r = i >> 2, c = (i & 3) << 3;
                copy_b128_to_lds(Abf + (u64)(m0 + r) * lda + (k0 + c), &sA[buf][r][c]);
            }
        } else {
            #pragma unroll
            for (int i = tid; i < 1024; i += 256) {          // 128 rows x 8 chunks(4 f32)
                int r = i >> 3, c = (i & 7) << 2;
                float4 v = *(const float4*)(Af + (u64)(m0 + r) * lda + (k0 + c));
                u32 lo = (u32)f2bf(v.x) | ((u32)f2bf(v.y) << 16);
                u32 hi = (u32)f2bf(v.z) | ((u32)f2bf(v.w) << 16);
                *(uint2*)&sA[buf][r][c] = make_uint2(lo, hi);
            }
        }
        #pragma unroll
        for (int i = tid; i < 512; i += 256) {               // B: 128 n-rows x 4 chunks
            int r = i >> 2, c = (i & 3) << 3;
            copy_b128_to_lds(Bp + (u64)(n0 + r) * ldb + (k0 + c), &sB[buf][r][c]);
        }
    };
    // async copies issued per thread per stage (only bf16 operands are async)
    constexpr int APS = (A_BF16 ? 2 : 0) + 2;

    const int nsteps = K >> 5;
    stage(0, 0);
    for (int s = 0; s < nsteps; ++s) {
        const int cur = s & 1;
        if (s + 1 < nsteps) { stage(cur ^ 1, (s + 1) << 5); wait_async<APS>(); }
        else                { wait_async<0>(); }
        __syncthreads();

        // ---- build fragments per CDNA5 16-bit WMMA VGPR layouts ----
        FragBF afr[4], bfr[2];
        #pragma unroll
        for (int mi = 0; mi < 4; ++mi) {
            const int row = wm + mi * 16 + l16;
            #pragma unroll
            for (int j = 0; j < 8; ++j) {
                const int e  = 2 * j;
                const int kk = ((e >> 3) << 4) + (half << 3) + (e & 7);
                afr[mi].u[j] = *(const u32*)&sA[cur][row][kk];
            }
        }
        #pragma unroll
        for (int ni = 0; ni < 2; ++ni) {
            const int col = wn + ni * 16 + l16;
            #pragma unroll
            for (int j = 0; j < 8; ++j)
                bfr[ni].u[j] = *(const u32*)&sB[cur][col][((j >> 2) << 4) + (half << 3) + 2 * (j & 3)];
        }

        #pragma unroll
        for (int mi = 0; mi < 4; ++mi)
            #pragma unroll
            for (int ni = 0; ni < 2; ++ni)
                acc[mi][ni] = __builtin_amdgcn_wmma_f32_16x16x32_bf16(
                    false, afr[mi].v, false, bfr[ni].v,
                    (short)0, acc[mi][ni], false, false);

        __syncthreads();   // protect buffer reuse (stage s+2 overwrites buf cur)
    }

    // ---- epilogue ----
    if (!OUT_BF16) {
        float* C = (float*)Cp;
        #pragma unroll
        for (int mi = 0; mi < 4; ++mi)
            #pragma unroll
            for (int ni = 0; ni < 2; ++ni)
                #pragma unroll
                for (int r = 0; r < 8; ++r) {
                    const int row = m0 + wm + mi * 16 + half * 8 + r;
                    const int col = n0 + wn + ni * 16 + l16;
                    C[(u64)row * ldc + col] = acc[mi][ni][r] * scale;
                }
    } else if (!OUT_TRANS) {
        u16* C = (u16*)Cp;
        #pragma unroll
        for (int mi = 0; mi < 4; ++mi)
            #pragma unroll
            for (int ni = 0; ni < 2; ++ni)
                #pragma unroll
                for (int r = 0; r < 8; ++r) {
                    const int row = m0 + wm + mi * 16 + half * 8 + r;
                    const int col = n0 + wn + ni * 16 + l16;
                    C[(u64)row * ldc + col] = f2bf(acc[mi][ni][r] * scale);
                }
    } else {
        // C^T[N,M] bf16: each lane owns 8 consecutive M -> one b128 store
        u16* C = (u16*)Cp;                                   // ldc = leading dim of C^T
        #pragma unroll
        for (int mi = 0; mi < 4; ++mi)
            #pragma unroll
            for (int ni = 0; ni < 2; ++ni) {
                const int ctr = n0 + wn + ni * 16 + l16;     // N index = C^T row
                const int ctc = m0 + wm + mi * 16 + half * 8;
                u32 pk[4];
                #pragma unroll
                for (int j = 0; j < 4; ++j)
                    pk[j] = (u32)f2bf(acc[mi][ni][2 * j] * scale) |
                            ((u32)f2bf(acc[mi][ni][2 * j + 1] * scale) << 16);
                *(uint4*)&C[(u64)ctr * ldc + ctc] = make_uint4(pk[0], pk[1], pk[2], pk[3]);
            }
    }
}

// ---------------------------------------------------------------------------
// f32 -> bf16 bulk convert (float4 in, 4x bf16 out per thread)
// ---------------------------------------------------------------------------
__global__ void cvt_bf16(const float* __restrict__ in, u16* __restrict__ out, long long n4)
{
    long long i = (long long)blockIdx.x * blockDim.x + threadIdx.x;
    if (i >= n4) return;
    float4 v = ((const float4*)in)[i];
    u32 lo = (u32)f2bf(v.x) | ((u32)f2bf(v.y) << 16);
    u32 hi = (u32)f2bf(v.z) | ((u32)f2bf(v.w) << 16);
    ((uint2*)out)[i] = make_uint2(lo, hi);
}

// ---------------------------------------------------------------------------
// Tiled transpose + convert: in[K,N] f32 -> out[N,K] bf16. K,N multiples of 32.
// ---------------------------------------------------------------------------
__global__ __launch_bounds__(256)
void transpose_bf16(const float* __restrict__ in, u16* __restrict__ out, int K, int N)
{
    __shared__ u16 t[32][33];
    const int k0 = blockIdx.y * 32, n0 = blockIdx.x * 32;
    const int tx = threadIdx.x & 31, ty = threadIdx.x >> 5;
    #pragma unroll
    for (int r = ty; r < 32; r += 8)
        t[r][tx] = f2bf(in[(u64)(k0 + r) * N + (n0 + tx)]);
    __syncthreads();
    #pragma unroll
    for (int r = ty; r < 32; r += 8)
        out[(u64)(n0 + r) * K + (k0 + tx)] = t[tx][r];
}

// ---------------------------------------------------------------------------
// RoPE on bf16 buffer (B,S,nheads,HD): pairs (d, d+HD/2) with cos/sin (B,S,HD)
// ---------------------------------------------------------------------------
__global__ void rope_bf16(u16* __restrict__ buf,
                          const float* __restrict__ cosp,
                          const float* __restrict__ sinp,
                          int nheads, long long total)
{
    long long idx = (long long)blockIdx.x * blockDim.x + threadIdx.x;
    if (idx >= total) return;
    const int half = HD_ / 2;
    int d = (int)(idx % half);
    long long t = idx / half;
    int h = (int)(t % nheads);
    long long bs = t / nheads;
    long long base = bs * (long long)nheads * HD_ + (long long)h * HD_;
    float c1 = cosp[bs * HD_ + d],        s1 = sinp[bs * HD_ + d];
    float c2 = cosp[bs * HD_ + d + half], s2 = sinp[bs * HD_ + d + half];
    float x1 = bf2f(buf[base + d]);
    float x2 = bf2f(buf[base + d + half]);
    buf[base + d]        = f2bf(x1 * c1 - x2 * s1);
    buf[base + d + half] = f2bf(x2 * c2 + x1 * s2);
}

// ---------------------------------------------------------------------------
// Row softmax over attn (B,H,S,S) f32 in place, adding mask (B,1,S,S).
// ---------------------------------------------------------------------------
__global__ __launch_bounds__(256)
void softmax_kernel(float* __restrict__ attn, const float* __restrict__ mask)
{
    const long long row = blockIdx.x;
    const int b = (int)(row / ((long long)H_ * S_));
    const int q = (int)(row % S_);
    float* p = attn + row * (long long)S_;
    const float* mrow = mask + ((long long)b * S_ + q) * (long long)S_;

    __shared__ float redmax[8];
    __shared__ float redsum[8];

    float m = -3.4e38f;
    for (int i = threadIdx.x; i < S_; i += 256) m = fmaxf(m, p[i] + mrow[i]);
    #pragma unroll
    for (int o = 16; o > 0; o >>= 1) m = fmaxf(m, __shfl_xor(m, o, 32));
    if ((threadIdx.x & 31) == 0) redmax[threadIdx.x >> 5] = m;
    __syncthreads();
    float bm = redmax[0];
    #pragma unroll
    for (int i = 1; i < 8; ++i) bm = fmaxf(bm, redmax[i]);

    float sum = 0.f;
    for (int i = threadIdx.x; i < S_; i += 256) {
        float v = __expf(p[i] + mrow[i] - bm);
        p[i] = v;
        sum += v;
    }
    #pragma unroll
    for (int o = 16; o > 0; o >>= 1) sum += __shfl_xor(sum, o, 32);
    if ((threadIdx.x & 31) == 0) redsum[threadIdx.x >> 5] = sum;
    __syncthreads();
    float bs = 0.f;
    #pragma unroll
    for (int i = 0; i < 8; ++i) bs += redsum[i];
    const float inv = 1.f / bs;
    for (int i = threadIdx.x; i < S_; i += 256) p[i] *= inv;
}

// ---------------------------------------------------------------------------
// Host driver
// ---------------------------------------------------------------------------
extern "C" void kernel_launch(void* const* d_in, const int* in_sizes, int n_in,
                              void* d_out, int out_size, void* d_ws, size_t ws_size,
                              hipStream_t stream)
{
    const float* hidden = (const float*)d_in[0];
    const float* mask   = (const float*)d_in[1];
    const float* cosp   = (const float*)d_in[2];
    const float* sinp   = (const float*)d_in[3];
    const float* Wq     = (const float*)d_in[4];   // (D, H*HD)
    const float* Wk     = (const float*)d_in[5];   // (D, HD)
    const float* Wv     = (const float*)d_in[6];   // (D, HD)
    const float* Wo     = (const float*)d_in[7];   // (H*HD, D)

    float* out  = (float*)d_out;                   // (B,S,D)
    float* attn = out + (u64)B_ * S_ * D_;         // (B,H,S,S) f32 (required output)

    // ---- bf16 workspace carve (~73 MB) ----
    u16* hbf   = (u16*)d_ws;                       // (BSR, D)
    u16* wqT   = hbf   + (u64)BSR * D_;            // (H*HD, D)
    u16* wkT   = wqT   + (u64)(H_ * HD_) * D_;     // (HD, D)
    u16* wvT   = wkT   + (u64)HD_ * D_;            // (HD, D)
    u16* woT   = wvT   + (u64)HD_ * D_;            // (D, H*HD)
    u16* qbf   = woT   + (u64)D_ * (H_ * HD_);     // (BSR, H*HD)
    u16* kbf   = qbf   + (u64)BSR * (H_ * HD_);    // (BSR, HD)
    u16* vTbf  = kbf   + (u64)BSR * HD_;           // (HD, BSR)   V^T
    u16* ctxbf = vTbf  + (u64)HD_ * BSR;           // (BSR, H*HD)

    const dim3 blk(256);

    // ---- one-time conversions / weight transposes ----
    cvt_bf16<<<(unsigned)(((u64)BSR * D_ / 4 + 255) / 256), blk, 0, stream>>>(
        hidden, hbf, (long long)BSR * D_ / 4);
    transpose_bf16<<<dim3((H_ * HD_) / 32, D_ / 32), blk, 0, stream>>>(Wq, wqT, D_, H_ * HD_);
    transpose_bf16<<<dim3(HD_ / 32, D_ / 32), blk, 0, stream>>>(Wk, wkT, D_, HD_);
    transpose_bf16<<<dim3(HD_ / 32, D_ / 32), blk, 0, stream>>>(Wv, wvT, D_, HD_);
    transpose_bf16<<<dim3(D_ / 32, (H_ * HD_) / 32), blk, 0, stream>>>(Wo, woT, H_ * HD_, D_);

    // ---- QKV projections (all NT, bf16 in/out; V written transposed) ----
    gemm_nt<true, true, false><<<dim3((H_ * HD_) / 128, BSR / 128), blk, 0, stream>>>(
        hbf, wqT, qbf, BSR, H_ * HD_, D_, D_, D_, H_ * HD_, 1.0f);
    gemm_nt<true, true, false><<<dim3(HD_ / 128, BSR / 128), blk, 0, stream>>>(
        hbf, wkT, kbf, BSR, HD_, D_, D_, D_, HD_, 1.0f);
    gemm_nt<true, true, true><<<dim3(HD_ / 128, BSR / 128), blk, 0, stream>>>(
        hbf, wvT, vTbf, BSR, HD_, D_, D_, D_, BSR /*ld of V^T*/, 1.0f);

    // ---- RoPE ----
    {
        long long totQ = (long long)BSR * H_ * (HD_ / 2);
        long long totK = (long long)BSR * HKV_ * (HD_ / 2);
        rope_bf16<<<(unsigned)((totQ + 255) / 256), blk, 0, stream>>>(qbf, cosp, sinp, H_, totQ);
        rope_bf16<<<(unsigned)((totK + 255) / 256), blk, 0, stream>>>(kbf, cosp, sinp, HKV_, totK);
    }

    // ---- scores = Q K^T / sqrt(HD) -> f32 attn output ----
    for (int bh = 0; bh < B_ * H_; ++bh) {
        int b = bh / H_, h = bh % H_;
        const u16* Apq = qbf + (u64)b * S_ * (H_ * HD_) + (u64)h * HD_;
        const u16* Bpk = kbf + (u64)b * S_ * HD_;
        float* Cp = attn + (u64)bh * S_ * S_;
        gemm_nt<true, false, false><<<dim3(S_ / 128, S_ / 128), blk, 0, stream>>>(
            Apq, Bpk, Cp, S_, S_, HD_, H_ * HD_, HD_, S_, 0.0625f);
    }

    // ---- softmax (adds mask), in place on f32 attn ----
    softmax_kernel<<<(unsigned)((u64)B_ * H_ * S_), blk, 0, stream>>>(attn, mask);

    // ---- ctx = attn @ V  (A = f32 attn, B = V^T bf16, NT) ----
    for (int bh = 0; bh < B_ * H_; ++bh) {
        int b = bh / H_, h = bh % H_;
        const float* Apa = attn + (u64)bh * S_ * S_;
        const u16* Bpv = vTbf + (u64)b * S_;          // column slice of V^T (ldb = BSR)
        u16* Cp = ctxbf + (u64)b * S_ * (H_ * HD_) + (u64)h * HD_;
        gemm_nt<false, true, false><<<dim3(HD_ / 128, S_ / 128), blk, 0, stream>>>(
            Apa, Bpv, Cp, S_, HD_, S_, S_, BSR, H_ * HD_, 1.0f);
    }

    // ---- output projection: out = ctx @ Wo  (B = Wo^T, NT) -> f32 ----
    gemm_nt<true, false, false><<<dim3(D_ / 128, BSR / 128), blk, 0, stream>>>(
        ctxbf, woT, out, BSR, D_, H_ * HD_, H_ * HD_, H_ * HD_, D_, 1.0f);
}
